// HolographicMixer_52561809768533
// MI455X (gfx1250) — matrix operand order repaired and checked by reading, not compiled, not run on hardware
//
#include <hip/hip_runtime.h>
#include <hip/hip_bf16.h>

// Problem constants (from reference)
#define B_  4
#define T_  2048
#define D_  1024
#define BT_ (B_ * T_)   // 8192 rows in the GEMM
#define TC  128         // scan chunk length
#define NC  (T_ / TC)   // 16 chunks

typedef __attribute__((ext_vector_type(16))) __bf16 v16bf;
typedef __attribute__((ext_vector_type(2)))  __bf16 v2bf;
typedef __attribute__((ext_vector_type(8)))  float  v8f;

__device__ __forceinline__ unsigned short f2bf(float f) {
  // round-to-nearest-even fp32 -> bf16 (cold-path use only)
  unsigned u = __float_as_uint(f);
  u += 0x7FFFu + ((u >> 16) & 1u);
  return (unsigned short)(u >> 16);
}

__device__ __forceinline__ float sigmoidf_(float x) {
  return 1.0f / (1.0f + __expf(-x));
}

// Wait for all of this wave's async (global<->LDS) ops to complete.
__device__ __forceinline__ void wait_async0() {
#if defined(__has_builtin) && __has_builtin(__builtin_amdgcn_s_wait_asynccnt)
  __builtin_amdgcn_s_wait_asynccnt(0);
#else
  asm volatile("s_wait_asynccnt 0x0" ::: "memory");
#endif
}

// Async copy 16B from global to LDS (ASYNCcnt-tracked; INST_OFFSET applies to
// both the LDS destination and the global source address).
__device__ __forceinline__ void async_b128(unsigned lds_off, const void* gaddr) {
  asm volatile("global_load_async_to_lds_b128 %0, %1, off"
               :: "v"(lds_off), "v"(gaddr) : "memory");
}
__device__ __forceinline__ void async_b128_o16(unsigned lds_off, const void* gaddr) {
  asm volatile("global_load_async_to_lds_b128 %0, %1, off offset:16"
               :: "v"(lds_off), "v"(gaddr) : "memory");
}

// Pack 8 fp32 (2x float4) into 8 bf16 (uint4) using native HW converts.
__device__ __forceinline__ uint4 pack8lo(float4 a, float4 b) {
  union { v2bf h2[4]; uint4 u; } p;
  p.h2[0] = (v2bf){(__bf16)a.x, (__bf16)a.y};
  p.h2[1] = (v2bf){(__bf16)a.z, (__bf16)a.w};
  p.h2[2] = (v2bf){(__bf16)b.x, (__bf16)b.y};
  p.h2[3] = (v2bf){(__bf16)b.z, (__bf16)b.w};
  return p.u;
}

// ---------------------------------------------------------------------------
// Kernel 1: combined kernel c = circ_corr(bind, unbind):
//   c[n] = sum_m bind[(n+m) mod D] * unbind[m]
// ---------------------------------------------------------------------------
__global__ __launch_bounds__(256)
void k_prep_c(const float* __restrict__ bind, const float* __restrict__ unbind,
              float* __restrict__ c) {
  __shared__ float sb[D_];
  __shared__ float su[D_];
  const int tid = threadIdx.x;
  for (int i = tid; i < D_; i += 256) { sb[i] = bind[i]; su[i] = unbind[i]; }
  __syncthreads();
  const int n = blockIdx.x * 256 + tid;
  float acc = 0.f;
#pragma unroll 4
  for (int m = 0; m < D_; ++m)
    acc = fmaf(sb[(n + m) & (D_ - 1)], su[m], acc);
  c[n] = acc;
}

// ---------------------------------------------------------------------------
// Kernel 2: materialize circulant weight, transposed for the GEMM B-operand:
//   Wt[n][k] = bf16( c[(n - k) mod D] )   (so z[m][n] = sum_k x[m][k]*Wt[n][k])
// ---------------------------------------------------------------------------
__global__ __launch_bounds__(256)
void k_prep_w(const float* __restrict__ c, unsigned short* __restrict__ Wt) {
  const int idx = blockIdx.x * 256 + threadIdx.x;     // 0 .. D*D-1
  const int n = idx >> 10;
  const int k = idx & (D_ - 1);
  Wt[idx] = f2bf(c[(n - k) & (D_ - 1)]);
}

// ---------------------------------------------------------------------------
// Kernel 3: WMMA GEMM  Z(8192x1024, f32) = X(f32, cast bf16) * W(bf16)
// Block: 256 threads = 8 wave32 waves, tile 128(M) x 128(N), K-step 32.
// Wave grid 4(M) x 2(N): each wave -> 2x4 grid of 16x16 f32 accumulators.
// Double-buffered LDS; B tile fetched with global_load_async_to_lds_b128,
// A tile register-staged (fp32 -> bf16 via native v_cvt) + ds_store_b128.
// One barrier per K-step.
// ---------------------------------------------------------------------------
#define TM 128
#define TN 128
#define TK 32
#define LDA 40   // padded LDS row stride (shorts): 80B = 20-bank stride, 16B aligned
#define LDB 40
#define BUFA (TM * LDA)   // shorts per A buffer (10240 B)
#define BUFB (TN * LDB)   // shorts per B buffer (10240 B)

__global__ __launch_bounds__(256)
void k_gemm(const float* __restrict__ X, const unsigned short* __restrict__ Wt,
            float* __restrict__ Z) {
  __shared__ alignas(16) unsigned short As[2][BUFA];
  __shared__ alignas(16) unsigned short Bs[2][BUFB];

  const int tid   = threadIdx.x;
  const int lane  = tid & 31;
  const int wid   = tid >> 5;
  const int waveM = wid & 3;    // 0..3 : 32-row band
  const int waveN = wid >> 2;   // 0..1 : 64-col band
  const int m0 = blockIdx.x * TM;
  const int n0 = blockIdx.y * TN;

  // cooperative staging indices: 2 threads per row, 16 elements (32B) each
  const int row  = tid >> 1;    // 0..127
  const int half = tid & 1;

  // fragment indices (ISA 7.12.2 layouts, wave32)
  const int fr   = lane & 15;
  const int hi16 = lane >> 4;   // 0: lanes 0-15, 1: lanes 16-31
  const int kA   = hi16 * 8;    // A: K {0..7,16..23} vs {8..15,24..31}
  const int kB   = hi16 * 8;    // B: K {0..7,16..23} vs {8..15,24..31}

  // per-thread staging addresses
  const float*          aSrc = X  + (size_t)(m0 + row) * D_ + half * 16;
  const unsigned short* bSrc = Wt + (size_t)(n0 + row) * D_ + half * 16;
  const unsigned ldsB0 = (unsigned)(uintptr_t)&Bs[0][row * LDB + half * 16];

  v8f acc[2][4] = {};

  // ---- prologue: fill buffer 0 with tile kb=0 ----
  {
    async_b128    (ldsB0, bSrc);
    async_b128_o16(ldsB0, bSrc);
    const float4* ap = (const float4*)aSrc;
    float4 f0 = ap[0], f1 = ap[1], f2 = ap[2], f3 = ap[3];
    *(uint4*)&As[0][row * LDA + half * 16]     = pack8lo(f0, f1);
    *(uint4*)&As[0][row * LDA + half * 16 + 8] = pack8lo(f2, f3);
    wait_async0();
    __syncthreads();
  }

  int cur = 0;
  for (int kb = 0; kb < D_; kb += TK) {
    const int  nxt     = cur ^ 1;
    const bool hasNext = (kb + TK) < D_;

    // ---- fragment loads from buffer `cur` ----
    v16bf afrag[2], bfrag[4];
#pragma unroll
    for (int i = 0; i < 2; ++i) {
      const unsigned short* s = &As[cur][(waveM * 32 + i * 16 + fr) * LDA + kA];
      union { uint4 u[2]; v16bf v; } cv;
      cv.u[0] = *(const uint4*)s;          // K = kA .. kA+7
      cv.u[1] = *(const uint4*)(s + 16);   // K = kA+16 .. kA+23
      afrag[i] = cv.v;
    }
#pragma unroll
    for (int j = 0; j < 4; ++j) {
      const unsigned short* s = &Bs[cur][(waveN * 64 + j * 16 + fr) * LDB + kB];
      union { uint4 u[2]; v16bf v; } cv;
      cv.u[0] = *(const uint4*)s;          // K = kB .. kB+7
      cv.u[1] = *(const uint4*)(s + 16);   // K = kB+16 .. kB+23
      bfrag[j] = cv.v;
    }

    // ---- kick off next tile while WMMAs run ----
    float4 f0, f1, f2, f3;
    if (hasNext) {
      const unsigned ldsBn = ldsB0 + (unsigned)(nxt * BUFB * 2);
      async_b128    (ldsBn, bSrc + (kb + TK));
      async_b128_o16(ldsBn, bSrc + (kb + TK));
      const float4* ap = (const float4*)(aSrc + (kb + TK));
      f0 = ap[0]; f1 = ap[1]; f2 = ap[2]; f3 = ap[3];
    }

    // ---- 8 WMMAs per wave per K-step ----
#pragma unroll
    for (int i = 0; i < 2; ++i)
#pragma unroll
      for (int j = 0; j < 4; ++j)
        acc[i][j] = __builtin_amdgcn_wmma_f32_16x16x32_bf16(
            false, afrag[i], false, bfrag[j], (short)0, acc[i][j], false, false);

    if (hasNext) {
      *(uint4*)&As[nxt][row * LDA + half * 16]     = pack8lo(f0, f1);
      *(uint4*)&As[nxt][row * LDA + half * 16 + 8] = pack8lo(f2, f3);
      wait_async0();              // B(nxt) landed in LDS
    }
    __syncthreads();              // publish buffer `nxt` (dscnt waited by codegen)
    cur = nxt;
  }

  // ---- epilogue: C/D layout — VGPR v holds M = sub + hi16*8 + v, N = sub + fr ----
  const int er = hi16 * 8;
#pragma unroll
  for (int i = 0; i < 2; ++i)
#pragma unroll
    for (int j = 0; j < 4; ++j) {
      const int m = m0 + waveM * 32 + i * 16 + er;
      const int n = n0 + waveN * 64 + j * 16 + fr;
      float* zp = Z + (size_t)m * D_ + n;
#pragma unroll
      for (int v = 0; v < 8; ++v)
        zp[(size_t)v * D_] = acc[i][j][v];
    }
}

// ---------------------------------------------------------------------------
// Kernel 4/5/6: chunked causal decay scan  h[t] = d*h[t-1] + z[t]
// ---------------------------------------------------------------------------
__global__ __launch_bounds__(256)
void k_carry(const float* __restrict__ Z, float* __restrict__ carry,
             const float* __restrict__ decay) {
  const float dd = sigmoidf_(decay[0]);
  const int d  = blockIdx.x * 256 + threadIdx.x;
  const int ci = blockIdx.y;
  const int b  = blockIdx.z;
  const float* zp = Z + (size_t)(b * T_ + ci * TC) * D_ + d;
  float r = 0.f;
#pragma unroll 4
  for (int t = 0; t < TC; ++t)
    r = fmaf(dd, r, zp[(size_t)t * D_]);
  carry[((size_t)b * NC + ci) * D_ + d] = r;
}

__global__ __launch_bounds__(256)
void k_chunkscan(const float* __restrict__ carry, float* __restrict__ hin,
                 const float* __restrict__ decay) {
  const float dd = sigmoidf_(decay[0]);
  const float w  = __powf(dd, (float)TC);
  const int idx = blockIdx.x * 256 + threadIdx.x;  // b*D + d
  const int b = idx >> 10;
  const int d = idx & (D_ - 1);
  float h = 0.f;
#pragma unroll
  for (int ci = 0; ci < NC; ++ci) {
    const size_t o = ((size_t)b * NC + ci) * D_ + d;
    hin[o] = h;
    h = fmaf(w, h, carry[o]);
  }
}

__global__ __launch_bounds__(256)
void k_apply(const float* __restrict__ Z, const float* __restrict__ X,
             const float* __restrict__ hin, float* __restrict__ out,
             const float* __restrict__ decay, const float* __restrict__ gate) {
  const float dd = sigmoidf_(decay[0]);
  const float g  = gate[0];
  const int d  = blockIdx.x * 256 + threadIdx.x;
  const int ci = blockIdx.y;
  const int b  = blockIdx.z;
  float r = hin[((size_t)b * NC + ci) * D_ + d];
  const size_t base = (size_t)(b * T_ + ci * TC) * D_ + d;
#pragma unroll 4
  for (int t = 0; t < TC; ++t) {
    const size_t o = base + (size_t)t * D_;
    r = fmaf(dd, r, Z[o]);
    out[o] = fmaf(g, r, X[o]);
  }
}

// ---------------------------------------------------------------------------
// Launcher
// ---------------------------------------------------------------------------
extern "C" void kernel_launch(void* const* d_in, const int* in_sizes, int n_in,
                              void* d_out, int out_size, void* d_ws, size_t ws_size,
                              hipStream_t stream) {
  const float* x      = (const float*)d_in[0];
  const float* bindk  = (const float*)d_in[1];
  const float* unbk   = (const float*)d_in[2];
  const float* gate   = (const float*)d_in[3];
  const float* decay  = (const float*)d_in[4];
  float* out = (float*)d_out;

  // workspace layout (bytes):
  //   [0,4096)                 c        : 1024 f32
  //   [4096,266240)            carry    : B*NC*D f32   (256 KB)
  //   [266240,528384)          hin      : B*NC*D f32   (256 KB)
  //   [528384,2625536)         Wt       : D*D bf16     (2 MB)
  //   [2625536,36180000)       Z        : B*T*D f32    (32 MB)
  char* ws = (char*)d_ws;
  float*          c     = (float*)(ws);
  float*          carry = (float*)(ws + 4096);
  float*          hin   = (float*)(ws + 4096 + 262144);
  unsigned short* Wt    = (unsigned short*)(ws + 528384);
  float*          Z     = (float*)(ws + 528384 + 2097152);

  k_prep_c<<<dim3(D_ / 256), 256, 0, stream>>>(bindk, unbk, c);
  k_prep_w<<<dim3((D_ * D_) / 256), 256, 0, stream>>>(c, Wt);

  k_gemm<<<dim3(BT_ / TM, D_ / TN), 256, 0, stream>>>(x, Wt, Z);

  k_carry<<<dim3(D_ / 256, NC, B_), 256, 0, stream>>>(Z, carry, decay);
  k_chunkscan<<<dim3((B_ * D_) / 256), 256, 0, stream>>>(carry, hin, decay);
  k_apply<<<dim3(D_ / 256, NC, B_), 256, 0, stream>>>(Z, x, hin, out, decay, gate);
}
